// GNNDetLSTM_13314398618217
// MI455X (gfx1250) — matrix operand order repaired and checked by reading, not compiled, not run on hardware
//
#include <hip/hip_runtime.h>
#include <hip/hip_bf16.h>

// ---------------- problem constants (match reference) ----------------
#define LNUM 12
#define NNODE 8192
#define ENUM 131072
#define DFEAT 768
#define GNUM 16
#define GH 128
#define LH 256

typedef __attribute__((ext_vector_type(16))) _Float16 v16h;
typedef __attribute__((ext_vector_type(8)))  float    v8f;

__device__ __forceinline__ void atomAddF(float* p, float v) {
    __hip_atomic_fetch_add(p, v, __ATOMIC_RELAXED, __HIP_MEMORY_SCOPE_AGENT);
}

// ---------------------------------------------------------------------
// WMMA GEMM, N fixed at 128: C[M,128] = act(A[M,K] @ W[K,128] + bias)
//
// Block = 256 threads (8 waves) -> block tile 64(M) x 128(N).
//   wave w: wm = w>>1 (M sub-tile), wn = w&1 (N half); wave tile 16x64,
//   4 v8f accumulators -> 4 v_wmma per K-step off a single A fragment.
// W is staged per 32-deep K-chunk into LDS, f32->f16, PRE-SWIZZLED into the
// contiguous-k B-fragment layout, as A uses (b[i] = B[k0+(lane>>4)*16+i][nt*16+(lane&15)])
// so each lane's fragment is 32 contiguous bytes -> 2x ds_load_b128.
// A fragments are gathered from global (rows are distinct per wave) with a
// global_prefetch on the next K-chunk.
// ---------------------------------------------------------------------
__global__ void gemm_wmma_f16(const float* __restrict__ A, int lda,
                              const float* __restrict__ W,
                              const float* __restrict__ bias,
                              float* __restrict__ C,
                              int M, int K, int doRelu) {
    __align__(32) __shared__ _Float16 sW[32 * 128];   // one K-chunk of W, swizzled

    const int lane = threadIdx.x & 31;
    const int wave = threadIdx.x >> 5;
    const int wm   = wave >> 1;          // 0..3  M sub-tile
    const int wn   = wave & 1;           // 0..1  N half (64 cols)
    const int hk   = lane >> 4;          // lane half
    const int l15  = lane & 15;

    const int rowbase = blockIdx.x * 64 + wm * 16;
    const int row     = rowbase + l15;   // A fragment row for this lane

    v8f acc0 = {}, acc1 = {}, acc2 = {}, acc3 = {};

    for (int k0 = 0; k0 < K; k0 += 32) {
        __syncthreads();
        // ---- cooperative stage: W[k0..k0+31][0..127] -> sW (swizzled f16) ----
        {
            const int base = threadIdx.x * 16;          // 4096 elems / 256 thr
#pragma unroll
            for (int j = 0; j < 16; ++j) {
                const int f  = base + j;
                const int kc = f >> 7;                  // 0..31
                const int n  = f & 127;                 // 0..127
                const int nt = n >> 4;                  // B tile 0..7
                const int ln = ((kc >> 4) << 4) | (n & 15);
                sW[(nt * 32 + ln) * 16 + (kc & 15)] =
                    (_Float16)W[(size_t)(k0 + kc) * 128 + n];
            }
        }
        __syncthreads();

        // ---- A fragment (16x32 f16) for this wave ----
        const float* Ap = A + (size_t)row * lda + k0 + hk * 16;
        if (k0 + 32 < K) __builtin_prefetch(Ap + 32, 0, 3);   // global_prefetch_b8
        v16h a;
#pragma unroll
        for (int i = 0; i < 8; ++i) a[i]     = (_Float16)Ap[i];
#pragma unroll
        for (int i = 0; i < 8; ++i) a[8 + i] = (_Float16)Ap[8 + i];

        // ---- 4 B fragments from LDS + 4 WMMAs ----
        const int ntb = wn * 4;
        const v16h* bp0 = reinterpret_cast<const v16h*>(&sW[((ntb + 0) * 32 + lane) * 16]);
        const v16h* bp1 = reinterpret_cast<const v16h*>(&sW[((ntb + 1) * 32 + lane) * 16]);
        const v16h* bp2 = reinterpret_cast<const v16h*>(&sW[((ntb + 2) * 32 + lane) * 16]);
        const v16h* bp3 = reinterpret_cast<const v16h*>(&sW[((ntb + 3) * 32 + lane) * 16]);
        acc0 = __builtin_amdgcn_wmma_f32_16x16x32_f16(false, a, false, *bp0, (short)0, acc0, false, false);
        acc1 = __builtin_amdgcn_wmma_f32_16x16x32_f16(false, a, false, *bp1, (short)0, acc1, false, false);
        acc2 = __builtin_amdgcn_wmma_f32_16x16x32_f16(false, a, false, *bp2, (short)0, acc2, false, false);
        acc3 = __builtin_amdgcn_wmma_f32_16x16x32_f16(false, a, false, *bp3, (short)0, acc3, false, false);
    }

    // ---- epilogue: C 16x16 f32 layout: VGPR r -> M=r+8*hk, N=lane&15 ----
    const int r0 = rowbase + (hk << 3);
    v8f accs[4] = {acc0, acc1, acc2, acc3};
#pragma unroll
    for (int nf = 0; nf < 4; ++nf) {
        const int col = wn * 64 + nf * 16 + l15;
        const float bb = bias[col];
#pragma unroll
        for (int r = 0; r < 8; ++r) {
            float v = accs[nf][r] + bb;
            if (doRelu) v = fmaxf(v, 0.0f);
            C[(size_t)(r0 + r) * 128 + col] = v;
        }
    }
}

// ---------------------------------------------------------------------
// GINE message + scatter:  hbuf[dst[e]] += relu(xin[src[e]] + clip(ea)*We + be)
// hbuf is pre-initialized with xin (fuses h = x + aggr). One block per edge.
// ---------------------------------------------------------------------
__global__ void edge_aggr(const float* __restrict__ xin,
                          const float* __restrict__ eattr,
                          const int* __restrict__ src,
                          const int* __restrict__ dst,
                          const float* __restrict__ We,
                          const float* __restrict__ be,
                          float* __restrict__ hbuf, int Dm) {
    const int e = blockIdx.x;
    const int s = src[e];
    const int t = dst[e];
    const float a = fminf(fmaxf(eattr[e], 0.0f), 1.0f);
    const float* xs = xin + (size_t)s * Dm;
    float* ht = hbuf + (size_t)t * Dm;
    for (int d = threadIdx.x; d < Dm; d += blockDim.x) {
        float v = fmaxf(xs[d] + a * We[d] + be[d], 0.0f);
        atomAddF(&ht[d], v);
    }
}

__global__ void count_nodes(const int* __restrict__ batch, float* __restrict__ counts, int n) {
    int i = blockIdx.x * blockDim.x + threadIdx.x;
    if (i < n) atomAddF(&counts[batch[i]], 1.0f);
}

__global__ void pool_accum(const float* __restrict__ h, const int* __restrict__ batch,
                           float* __restrict__ pooled_l, int n, int dm) {
    int i = blockIdx.x * blockDim.x + threadIdx.x;
    if (i < n * dm) {
        int node = i / dm;
        int d = i - node * dm;
        atomAddF(&pooled_l[(size_t)batch[node] * dm + d], h[i]);
    }
}

__global__ void pool_div(float* __restrict__ pooled, const float* __restrict__ counts, int total) {
    int i = blockIdx.x * blockDim.x + threadIdx.x;
    if (i < total) {
        int g = (i / GH) % GNUM;
        pooled[i] /= fmaxf(counts[g], 1.0f);
    }
}

// ---------------------------------------------------------------------
// LSTM single step (one direction). block = (b in 0..15), thread = unit u.
// PyTorch gate order i,f,g,o as rows u, 256+u, 512+u, 768+u.
// ---------------------------------------------------------------------
__device__ __forceinline__ float sigf(float x) { return 1.0f / (1.0f + expf(-x)); }

__global__ void lstm_step(const float* __restrict__ xt, int Kin,
                          const float* __restrict__ Wih,
                          const float* __restrict__ Whh,
                          const float* __restrict__ bih,
                          const float* __restrict__ bhh,
                          const float* __restrict__ h_prev,
                          const float* __restrict__ c_prev,
                          float* __restrict__ h_new,
                          float* __restrict__ c_new,
                          float* __restrict__ y_out, int ystride) {
    const int b = blockIdx.x;
    const int u = threadIdx.x;            // 0..255
    const float* xb = xt + (size_t)b * Kin;
    const float* hb = h_prev + (size_t)b * LH;

    float gate[4];
#pragma unroll
    for (int gi = 0; gi < 4; ++gi) {
        const int r = gi * LH + u;
        float acc = bih[r] + bhh[r];
        const float* wih = Wih + (size_t)r * Kin;
        for (int k = 0; k < Kin; ++k) acc += xb[k] * wih[k];
        const float* whh = Whh + (size_t)r * LH;
        for (int k = 0; k < LH; ++k) acc += hb[k] * whh[k];
        gate[gi] = acc;
    }
    float c = sigf(gate[1]) * c_prev[(size_t)b * LH + u] + sigf(gate[0]) * tanhf(gate[2]);
    float h = sigf(gate[3]) * tanhf(c);
    c_new[(size_t)b * LH + u] = c;
    h_new[(size_t)b * LH + u] = h;
    if (y_out) y_out[(size_t)b * ystride + u] = h;
}

// ---------------------------------------------------------------------
// MLP head: out[b] = (relu(relu(cat(hf,hb) @ W1 + b1) @ W2 + b2)) @ W3 + b3
// ---------------------------------------------------------------------
__global__ void fc_head(const float* __restrict__ hf, const float* __restrict__ hb,
                        const float* __restrict__ W1, const float* __restrict__ b1,
                        const float* __restrict__ W2, const float* __restrict__ b2,
                        const float* __restrict__ W3, const float* __restrict__ b3,
                        float* __restrict__ out) {
    const int b = blockIdx.x;       // 0..15
    const int u = threadIdx.x;      // 0..127
    __shared__ float s1[128];
    __shared__ float s2[64];
    float acc = b1[u];
    const float* f = hf + (size_t)b * LH;
    const float* r = hb + (size_t)b * LH;
    for (int k = 0; k < LH; ++k) acc += f[k] * W1[(size_t)k * 128 + u];
    for (int k = 0; k < LH; ++k) acc += r[k] * W1[(size_t)(LH + k) * 128 + u];
    s1[u] = fmaxf(acc, 0.0f);
    __syncthreads();
    if (u < 64) {
        float a2 = b2[u];
        for (int k = 0; k < 128; ++k) a2 += s1[k] * W2[(size_t)k * 64 + u];
        s2[u] = fmaxf(a2, 0.0f);
    }
    __syncthreads();
    if (u == 0) {
        float a3 = b3[0];
        for (int k = 0; k < 64; ++k) a3 += s2[k] * W3[k];
        out[b] = a3;
    }
}

// ---------------------------------------------------------------------
extern "C" void kernel_launch(void* const* d_in, const int* in_sizes, int n_in,
                              void* d_out, int out_size, void* d_ws, size_t ws_size,
                              hipStream_t stream) {
    // inputs in setup_inputs() dict order
    const float* x         = (const float*)d_in[0];   // [L,N,D]
    const float* edge_attr = (const float*)d_in[1];   // [L,E,1]
    const int*   edge_idx  = (const int*)  d_in[2];   // [2,E]
    const int*   batch     = (const int*)  d_in[3];   // [N]
    const float* We1 = (const float*)d_in[4];  const float* be1 = (const float*)d_in[5];
    const float* W1a = (const float*)d_in[6];  const float* b1a = (const float*)d_in[7];
    const float* W1b = (const float*)d_in[8];  const float* b1b = (const float*)d_in[9];
    const float* We2 = (const float*)d_in[10]; const float* be2 = (const float*)d_in[11];
    const float* W2a = (const float*)d_in[12]; const float* b2a = (const float*)d_in[13];
    const float* W2b = (const float*)d_in[14]; const float* b2b = (const float*)d_in[15];
    const float* Wih0 = (const float*)d_in[16]; const float* Whh0 = (const float*)d_in[17];
    const float* bih0 = (const float*)d_in[18]; const float* bhh0 = (const float*)d_in[19];
    const float* Wih1 = (const float*)d_in[20]; const float* Whh1 = (const float*)d_in[21];
    const float* bih1 = (const float*)d_in[22]; const float* bhh1 = (const float*)d_in[23];
    const float* fc1_W = (const float*)d_in[24]; const float* fc1_b = (const float*)d_in[25];
    const float* fc2_W = (const float*)d_in[26]; const float* fc2_b = (const float*)d_in[27];
    const float* fc3_W = (const float*)d_in[28]; const float* fc3_b = (const float*)d_in[29];
    float* out = (float*)d_out;

    const int* src = edge_idx;          // edge_index[0]
    const int* dst = edge_idx + ENUM;   // edge_index[1]

    // ---- workspace carve ----
    char* p = (char*)d_ws;
    auto take = [&](size_t bytes) { char* q = p; p += (bytes + 255) & ~size_t(255); return q; };
    float* hbuf1   = (float*)take((size_t)NNODE * DFEAT * 4);   // x + aggr (768)
    float* t1      = (float*)take((size_t)NNODE * GH * 4);
    float* g1      = (float*)take((size_t)NNODE * GH * 4);      // conv1 output
    float* hbuf2   = (float*)take((size_t)NNODE * GH * 4);
    float* t2      = (float*)take((size_t)NNODE * GH * 4);
    float* g2      = (float*)take((size_t)NNODE * GH * 4);      // conv2 output
    float* pooled  = (float*)take((size_t)LNUM * GNUM * GH * 4);
    float* counts  = (float*)take((size_t)GNUM * 4);
    float* y1      = (float*)take((size_t)LNUM * GNUM * 2 * LH * 4);
    float* hs0     = (float*)take((size_t)GNUM * LH * 4);
    float* hs1     = (float*)take((size_t)GNUM * LH * 4);
    float* cs0     = (float*)take((size_t)GNUM * LH * 4);
    float* cs1     = (float*)take((size_t)GNUM * LH * 4);
    float* hf_fin  = (float*)take((size_t)GNUM * LH * 4);
    float* hb_fin  = (float*)take((size_t)GNUM * LH * 4);
    float* hsP[2] = {hs0, hs1};
    float* csP[2] = {cs0, cs1};

    // ---- graph counts + pooled init ----
    hipMemsetAsync(counts, 0, GNUM * 4, stream);
    hipMemsetAsync(pooled, 0, (size_t)LNUM * GNUM * GH * 4, stream);
    count_nodes<<<(NNODE + 255) / 256, 256, 0, stream>>>(batch, counts, NNODE);

    // GEMM geometry: block tile 64x128, 256 threads
    const int gblocks = NNODE / 64;   // 128

    // ---- 12 GNN layers ----
    for (int l = 0; l < LNUM; ++l) {
        const float* xl  = x + (size_t)l * NNODE * DFEAT;
        const float* eal = edge_attr + (size_t)l * ENUM;

        // conv1: hbuf1 = x + sum relu(x[src] + lin_edge(e))
        hipMemcpyAsync(hbuf1, xl, (size_t)NNODE * DFEAT * 4, hipMemcpyDeviceToDevice, stream);
        edge_aggr<<<ENUM, 256, 0, stream>>>(xl, eal, src, dst, We1, be1, hbuf1, DFEAT);
        // t1 = relu(hbuf1 @ W1a + b1a); g1 = relu(t1 @ W1b + b1b)  (outer relu of layer)
        gemm_wmma_f16<<<gblocks, 256, 0, stream>>>(hbuf1, DFEAT, W1a, b1a, t1, NNODE, DFEAT, 1);
        gemm_wmma_f16<<<gblocks, 256, 0, stream>>>(t1, GH, W1b, b1b, g1, NNODE, GH, 1);

        // conv2 (no outer relu)
        hipMemcpyAsync(hbuf2, g1, (size_t)NNODE * GH * 4, hipMemcpyDeviceToDevice, stream);
        edge_aggr<<<ENUM, 128, 0, stream>>>(g1, eal, src, dst, We2, be2, hbuf2, GH);
        gemm_wmma_f16<<<gblocks, 256, 0, stream>>>(hbuf2, GH, W2a, b2a, t2, NNODE, GH, 1);
        gemm_wmma_f16<<<gblocks, 256, 0, stream>>>(t2, GH, W2b, b2b, g2, NNODE, GH, 0);

        // mean-pool accumulate
        pool_accum<<<(NNODE * GH + 255) / 256, 256, 0, stream>>>(
            g2, batch, pooled + (size_t)l * GNUM * GH, NNODE, GH);
    }
    pool_div<<<(LNUM * GNUM * GH + 255) / 256, 256, 0, stream>>>(pooled, counts,
                                                                 LNUM * GNUM * GH);

    // ---- BiLSTM layer 0: GH -> LH, outputs full sequence y1 [T,G,2*LH] ----
    for (int dir = 0; dir < 2; ++dir) {
        const float* Wih = Wih0 + (size_t)dir * 4 * LH * GH;
        const float* Whh = Whh0 + (size_t)dir * 4 * LH * LH;
        const float* bi  = bih0 + (size_t)dir * 4 * LH;
        const float* bh  = bhh0 + (size_t)dir * 4 * LH;
        hipMemsetAsync(hsP[0], 0, (size_t)GNUM * LH * 4, stream);
        hipMemsetAsync(csP[0], 0, (size_t)GNUM * LH * 4, stream);
        for (int s = 0; s < LNUM; ++s) {
            const int t = dir ? (LNUM - 1 - s) : s;
            lstm_step<<<GNUM, LH, 0, stream>>>(
                pooled + (size_t)t * GNUM * GH, GH, Wih, Whh, bi, bh,
                hsP[s & 1], csP[s & 1], hsP[(s + 1) & 1], csP[(s + 1) & 1],
                y1 + (size_t)t * GNUM * 2 * LH + (size_t)dir * LH, 2 * LH);
        }
    }

    // ---- BiLSTM layer 1: 2*LH -> LH, keep only final hidden states ----
    for (int dir = 0; dir < 2; ++dir) {
        const float* Wih = Wih1 + (size_t)dir * 4 * LH * (2 * LH);
        const float* Whh = Whh1 + (size_t)dir * 4 * LH * LH;
        const float* bi  = bih1 + (size_t)dir * 4 * LH;
        const float* bh  = bhh1 + (size_t)dir * 4 * LH;
        hipMemsetAsync(hsP[0], 0, (size_t)GNUM * LH * 4, stream);
        hipMemsetAsync(csP[0], 0, (size_t)GNUM * LH * 4, stream);
        for (int s = 0; s < LNUM; ++s) {
            const int t = dir ? (LNUM - 1 - s) : s;
            lstm_step<<<GNUM, LH, 0, stream>>>(
                y1 + (size_t)t * GNUM * 2 * LH, 2 * LH, Wih, Whh, bi, bh,
                hsP[s & 1], csP[s & 1], hsP[(s + 1) & 1], csP[(s + 1) & 1],
                nullptr, 0);
        }
        // after s=11 the carry lives in slot 0
        hipMemcpyAsync(dir ? hb_fin : hf_fin, hsP[0], (size_t)GNUM * LH * 4,
                       hipMemcpyDeviceToDevice, stream);
    }

    // ---- MLP head -> out [16] ----
    fc_head<<<GNUM, 128, 0, stream>>>(hf_fin, hb_fin, fc1_W, fc1_b, fc2_W, fc2_b,
                                      fc3_W, fc3_b, out);
}
